// LlamaMlpWithLora_28286654612000
// MI455X (gfx1250) — matrix operand order repaired and checked by reading, not compiled, not run on hardware
//
#include <hip/hip_runtime.h>
#include <hip/hip_bf16.h>
#include <math.h>

typedef __attribute__((ext_vector_type(16))) __bf16        v16bf;
typedef __attribute__((ext_vector_type(8)))  __bf16        v8bf;
typedef __attribute__((ext_vector_type(4)))  __bf16        v4bf;
typedef __attribute__((ext_vector_type(8)))  float         v8f;
typedef __attribute__((ext_vector_type(16))) short         v16s;
typedef __attribute__((ext_vector_type(8)))  short         v8s;

#define T_TOK 4096
#define HID   4096
#define INTER 11008
#define RANK  16
#define NADP  4
#define AR    64            // NADP * RANK
#define LDK   40            // padded LDS row stride (elements) for a 32-K tile

// ---------------- helpers ----------------

static __device__ inline v8f vzero8() {
  v8f z = {0.f, 0.f, 0.f, 0.f, 0.f, 0.f, 0.f, 0.f};
  return z;
}

static __device__ inline v8f wmma_bf16(v16bf a, v16bf b, v8f c) {
  return __builtin_amdgcn_wmma_f32_16x16x32_bf16(
      /*neg_a=*/false, a, /*neg_b=*/false, b,
      /*c_mod=*/(short)0, c, /*reuse_a=*/false, /*reuse_b=*/false);
}

// A fragment (16x32 bf16): lane<16 holds K 0..15 contiguous, lane>=16 holds
// K 16..31.  rowk = &A[row(lane)*stride + k0]
static __device__ inline v16bf ldA(const unsigned short* __restrict__ rowk, int lane) {
  const int ks = (lane >> 4) << 4;  // 0 or 16
  v8s c0 = *(const v8s*)(rowk + ks);
  v8s c1 = *(const v8s*)(rowk + ks + 8);
  v16s s = __builtin_shufflevector(c0, c1, 0, 1, 2, 3, 4, 5, 6, 7,
                                   8, 9, 10, 11, 12, 13, 14, 15);
  return __builtin_bit_cast(v16bf, s);
}

// B fragment (32x16 bf16): lane<16 holds K 0..15 contiguous, lane>=16 holds K 16..31.
static __device__ inline v16bf ldB_bf(const unsigned short* __restrict__ rowk, int lane) {
  const v16s* p = (const v16s*)(rowk + ((lane >> 4) << 4));
  return __builtin_bit_cast(v16bf, *p);
}

// Convert 8 consecutive fp32 -> 8 bf16 (packed converts).
static __device__ inline v8s cvt8(const float* __restrict__ p) {
  float4 f0 = *(const float4*)(p + 0);
  float4 f1 = *(const float4*)(p + 4);
  v8bf r;
  r[0] = (__bf16)f0.x; r[1] = (__bf16)f0.y; r[2] = (__bf16)f0.z; r[3] = (__bf16)f0.w;
  r[4] = (__bf16)f1.x; r[5] = (__bf16)f1.y; r[6] = (__bf16)f1.z; r[7] = (__bf16)f1.w;
  return __builtin_bit_cast(v8s, r);
}

// ---------------- kernel 1: x fp32 -> bf16 ----------------

__global__ __launch_bounds__(256) void cvt_x_kernel(const float* __restrict__ x,
                                                    unsigned short* __restrict__ xb) {
  size_t idx = ((size_t)blockIdx.x * 256 + threadIdx.x) * 4;
  float4 v = *(const float4*)(x + idx);
  v4bf o;
  o[0] = (__bf16)v.x; o[1] = (__bf16)v.y; o[2] = (__bf16)v.z; o[3] = (__bf16)v.w;
  *(v4bf*)(xb + idx) = o;
}

// ---------------- kernel 2: wb [A*R, N] -> wbt [N, 64] bf16 ----------------

__global__ __launch_bounds__(256) void wbt_kernel(const float* __restrict__ src,
                                                  unsigned short* __restrict__ dst, int n) {
  int idx = blockIdx.x * 256 + threadIdx.x;  // n * 64 total
  int i = idx >> 6;
  int j = idx & 63;
  ((__bf16*)dst)[(size_t)i * AR + j] = (__bf16)src[(size_t)j * n + i];
}

// ---------------- kernel 3: per-token LoRA projection for gate & up ----------------

__global__ __launch_bounds__(256) void lora_proj_gu_kernel(
    const float* __restrict__ x, const float* __restrict__ gate_wa,
    const float* __restrict__ up_wa, const int* __restrict__ seg_ids,
    unsigned short* __restrict__ xg, unsigned short* __restrict__ xu) {
  const int t = blockIdx.x;
  const int tid = threadIdx.x;
  const int lane = tid & 31;
  const int wave = tid >> 5;
  const int a = seg_ids[t];

  const float* xr = x + (size_t)t * HID;
  const float* gw = gate_wa + (size_t)a * HID * RANK;
  const float* uw = up_wa + (size_t)a * HID * RANK;

  float ag[RANK], au[RANK];
#pragma unroll
  for (int r = 0; r < RANK; r++) { ag[r] = 0.f; au[r] = 0.f; }

  for (int h = tid; h < HID; h += 256) {
    float xv = xr[h];
    const float* g = gw + (size_t)h * RANK;
    const float* u = uw + (size_t)h * RANK;
#pragma unroll
    for (int r = 0; r < RANK; r++) { ag[r] += xv * g[r]; au[r] += xv * u[r]; }
  }
#pragma unroll
  for (int r = 0; r < RANK; r++) {
#pragma unroll
    for (int off = 16; off > 0; off >>= 1) {
      ag[r] += __shfl_down(ag[r], off, 32);
      au[r] += __shfl_down(au[r], off, 32);
    }
  }
  __shared__ float redg[8][RANK];
  __shared__ float redu[8][RANK];
  __shared__ float fing[RANK];
  __shared__ float finu[RANK];
  if (lane == 0) {
#pragma unroll
    for (int r = 0; r < RANK; r++) { redg[wave][r] = ag[r]; redu[wave][r] = au[r]; }
  }
  __syncthreads();
  if (tid < RANK) {
    float sg = 0.f, su = 0.f;
#pragma unroll
    for (int w = 0; w < 8; w++) { sg += redg[w][tid]; su += redu[w][tid]; }
    fing[tid] = sg; finu[tid] = su;
  }
  __syncthreads();
  if (tid < AR) {
    int adap = tid >> 4, r = tid & 15;
    ((__bf16*)xg)[(size_t)t * AR + tid] = (adap == a) ? (__bf16)fing[r] : (__bf16)0.0f;
    ((__bf16*)xu)[(size_t)t * AR + tid] = (adap == a) ? (__bf16)finu[r] : (__bf16)0.0f;
  }
}

// ---------------- kernel 4: fused gate/up WMMA GEMM + LoRA + silu*up -> t ----------------
// BM=128, BN=64, 8 waves as 4(M) x 2(N), each wave 2x2 WMMA tiles of 16x16.
// A and both weight tiles are staged through LDS (converted to bf16 once per block).

__global__ __launch_bounds__(256) void gemm_gate_up_kernel(
    const unsigned short* __restrict__ xb, const float* __restrict__ gate_w,
    const float* __restrict__ up_w, const unsigned short* __restrict__ xg,
    const unsigned short* __restrict__ xu, const unsigned short* __restrict__ gwbt,
    const unsigned short* __restrict__ uwbt, unsigned short* __restrict__ tmat) {
  __shared__ __align__(16) unsigned short lsA[128 * LDK];
  __shared__ __align__(16) unsigned short lsG[64 * LDK];
  __shared__ __align__(16) unsigned short lsU[64 * LDK];

  const int tid = threadIdx.x;
  const int lane = tid & 31;
  const int wave = tid >> 5;
  const int wm = wave & 3;   // 4 waves along M
  const int wn = wave >> 2;  // 2 waves along N
  const int lr = lane & 15;

  const int rowblk = blockIdx.x * 128;
  const int colblk = blockIdx.y * 64;
  const int row0 = wm * 32;  // within block
  const int col0 = wn * 32;

  // staging assignments: A covers 128x32 bf16, 16 elems/thread; weights 64x32 fp32, 8/thread
  const int arow = tid >> 1;        // 0..127
  const int akp = (tid & 1) * 16;   // 0 or 16
  const unsigned short* agp = xb + (size_t)(rowblk + arow) * HID + akp;

  const int wrow = tid >> 2;        // 0..63
  const int wkp = (tid & 3) * 8;    // 0,8,16,24
  const float* ggp = gate_w + (size_t)(colblk + wrow) * HID + wkp;
  const float* ugp = up_w + (size_t)(colblk + wrow) * HID + wkp;

  v8f accg[2][2], accu[2][2];
#pragma unroll
  for (int i = 0; i < 2; i++)
#pragma unroll
    for (int j = 0; j < 2; j++) { accg[i][j] = vzero8(); accu[i][j] = vzero8(); }

  const int NKB = HID / 32;
  v16s aS = *(const v16s*)agp;
  v8s gS = cvt8(ggp);
  v8s uS = cvt8(ugp);

  for (int kb = 0; kb < NKB; ++kb) {
    __syncthreads();  // previous tile fully consumed
    *(v16s*)(lsA + arow * LDK + akp) = aS;
    *(v8s*)(lsG + wrow * LDK + wkp) = gS;
    *(v8s*)(lsU + wrow * LDK + wkp) = uS;
    __syncthreads();  // tile visible

    if (kb + 1 < NKB) {  // fetch next tile into registers while WMMAs run
      const int k = (kb + 1) * 32;
      __builtin_prefetch(ggp + k + 256, 0, 3);
      __builtin_prefetch(ugp + k + 256, 0, 3);
      aS = *(const v16s*)(agp + k);
      gS = cvt8(ggp + k);
      uS = cvt8(ugp + k);
    }

    v16bf a0 = ldA(lsA + (row0 + lr) * LDK, lane);
    v16bf a1 = ldA(lsA + (row0 + 16 + lr) * LDK, lane);
    v16bf bg0 = ldB_bf(lsG + (col0 + lr) * LDK, lane);
    v16bf bg1 = ldB_bf(lsG + (col0 + 16 + lr) * LDK, lane);
    v16bf bu0 = ldB_bf(lsU + (col0 + lr) * LDK, lane);
    v16bf bu1 = ldB_bf(lsU + (col0 + 16 + lr) * LDK, lane);
    accg[0][0] = wmma_bf16(a0, bg0, accg[0][0]);
    accg[0][1] = wmma_bf16(a0, bg1, accg[0][1]);
    accg[1][0] = wmma_bf16(a1, bg0, accg[1][0]);
    accg[1][1] = wmma_bf16(a1, bg1, accg[1][1]);
    accu[0][0] = wmma_bf16(a0, bu0, accu[0][0]);
    accu[0][1] = wmma_bf16(a0, bu1, accu[0][1]);
    accu[1][0] = wmma_bf16(a1, bu0, accu[1][0]);
    accu[1][1] = wmma_bf16(a1, bu1, accu[1][1]);
  }

  // augmented K: += xa @ wb_t^T over the 64 (A*R) slots (tiny; direct from global)
  const unsigned short* gx0 = xg + (size_t)(rowblk + row0 + lr) * AR;
  const unsigned short* gx1 = xg + (size_t)(rowblk + row0 + 16 + lr) * AR;
  const unsigned short* ux0 = xu + (size_t)(rowblk + row0 + lr) * AR;
  const unsigned short* ux1 = xu + (size_t)(rowblk + row0 + 16 + lr) * AR;
  const unsigned short* gb0 = gwbt + (size_t)(colblk + col0 + lr) * AR;
  const unsigned short* gb1 = gwbt + (size_t)(colblk + col0 + 16 + lr) * AR;
  const unsigned short* ub0 = uwbt + (size_t)(colblk + col0 + lr) * AR;
  const unsigned short* ub1 = uwbt + (size_t)(colblk + col0 + 16 + lr) * AR;
#pragma unroll
  for (int k = 0; k < AR; k += 32) {
    v16bf a0 = ldA(gx0 + k, lane);
    v16bf a1 = ldA(gx1 + k, lane);
    v16bf b0 = ldB_bf(gb0 + k, lane);
    v16bf b1 = ldB_bf(gb1 + k, lane);
    accg[0][0] = wmma_bf16(a0, b0, accg[0][0]);
    accg[0][1] = wmma_bf16(a0, b1, accg[0][1]);
    accg[1][0] = wmma_bf16(a1, b0, accg[1][0]);
    accg[1][1] = wmma_bf16(a1, b1, accg[1][1]);
    v16bf c0 = ldA(ux0 + k, lane);
    v16bf c1 = ldA(ux1 + k, lane);
    v16bf d0 = ldB_bf(ub0 + k, lane);
    v16bf d1 = ldB_bf(ub1 + k, lane);
    accu[0][0] = wmma_bf16(c0, d0, accu[0][0]);
    accu[0][1] = wmma_bf16(c0, d1, accu[0][1]);
    accu[1][0] = wmma_bf16(c1, d0, accu[1][0]);
    accu[1][1] = wmma_bf16(c1, d1, accu[1][1]);
  }

  // epilogue: t = silu(gate) * up, stored bf16
  const int rbase = (lane >> 4) << 3;  // lanes 16..31 -> rows +8
  __bf16* tb = (__bf16*)tmat;
#pragma unroll
  for (int mt = 0; mt < 2; mt++)
#pragma unroll
    for (int nt = 0; nt < 2; nt++)
#pragma unroll
      for (int v = 0; v < 8; v++) {
        int rr = rowblk + row0 + mt * 16 + rbase + v;
        int cc = colblk + col0 + nt * 16 + lr;
        float g = accg[mt][nt][v];
        float u = accu[mt][nt][v];
        float s = g * (1.0f / (1.0f + __expf(-g)));
        tb[(size_t)rr * INTER + cc] = (__bf16)(s * u);
      }
}

// ---------------- kernel 5: per-token LoRA projection for down (on t) ----------------

__global__ __launch_bounds__(256) void lora_proj_down_kernel(
    const unsigned short* __restrict__ tmat, const float* __restrict__ down_wa,
    const int* __restrict__ seg_ids, unsigned short* __restrict__ ta) {
  const int t = blockIdx.x;
  const int tid = threadIdx.x;
  const int lane = tid & 31;
  const int wave = tid >> 5;
  const int a = seg_ids[t];

  const __bf16* tr = (const __bf16*)(tmat + (size_t)t * INTER);
  const float* w = down_wa + (size_t)a * INTER * RANK;

  float acc[RANK];
#pragma unroll
  for (int r = 0; r < RANK; r++) acc[r] = 0.f;

  for (int i = tid; i < INTER; i += 256) {
    float tv = (float)tr[i];
    const float* p = w + (size_t)i * RANK;
#pragma unroll
    for (int r = 0; r < RANK; r++) acc[r] += tv * p[r];
  }
#pragma unroll
  for (int r = 0; r < RANK; r++) {
#pragma unroll
    for (int off = 16; off > 0; off >>= 1) acc[r] += __shfl_down(acc[r], off, 32);
  }
  __shared__ float red[8][RANK];
  __shared__ float fin[RANK];
  if (lane == 0) {
#pragma unroll
    for (int r = 0; r < RANK; r++) red[wave][r] = acc[r];
  }
  __syncthreads();
  if (tid < RANK) {
    float s = 0.f;
#pragma unroll
    for (int w8 = 0; w8 < 8; w8++) s += red[w8][tid];
    fin[tid] = s;
  }
  __syncthreads();
  if (tid < AR) {
    int adap = tid >> 4, r = tid & 15;
    ((__bf16*)ta)[(size_t)t * AR + tid] = (adap == a) ? (__bf16)fin[r] : (__bf16)0.0f;
  }
}

// ---------------- kernel 6: down WMMA GEMM + LoRA -> out (fp32) ----------------

__global__ __launch_bounds__(256) void gemm_down_kernel(
    const unsigned short* __restrict__ tmat, const float* __restrict__ down_w,
    const unsigned short* __restrict__ ta, const unsigned short* __restrict__ dwbt,
    float* __restrict__ out) {
  __shared__ __align__(16) unsigned short lsA[128 * LDK];
  __shared__ __align__(16) unsigned short lsW[64 * LDK];

  const int tid = threadIdx.x;
  const int lane = tid & 31;
  const int wave = tid >> 5;
  const int wm = wave & 3;
  const int wn = wave >> 2;
  const int lr = lane & 15;

  const int rowblk = blockIdx.x * 128;
  const int colblk = blockIdx.y * 64;
  const int row0 = wm * 32;
  const int col0 = wn * 32;

  const int arow = tid >> 1;
  const int akp = (tid & 1) * 16;
  const unsigned short* agp = tmat + (size_t)(rowblk + arow) * INTER + akp;

  const int wrow = tid >> 2;
  const int wkp = (tid & 3) * 8;
  const float* wgp = down_w + (size_t)(colblk + wrow) * INTER + wkp;

  v8f acc[2][2];
#pragma unroll
  for (int i = 0; i < 2; i++)
#pragma unroll
    for (int j = 0; j < 2; j++) acc[i][j] = vzero8();

  const int NKB = INTER / 32;
  v16s aS = *(const v16s*)agp;
  v8s wS = cvt8(wgp);

  for (int kb = 0; kb < NKB; ++kb) {
    __syncthreads();
    *(v16s*)(lsA + arow * LDK + akp) = aS;
    *(v8s*)(lsW + wrow * LDK + wkp) = wS;
    __syncthreads();

    if (kb + 1 < NKB) {
      const int k = (kb + 1) * 32;
      __builtin_prefetch(wgp + k + 256, 0, 3);
      aS = *(const v16s*)(agp + k);
      wS = cvt8(wgp + k);
    }

    v16bf a0 = ldA(lsA + (row0 + lr) * LDK, lane);
    v16bf a1 = ldA(lsA + (row0 + 16 + lr) * LDK, lane);
    v16bf b0 = ldB_bf(lsW + (col0 + lr) * LDK, lane);
    v16bf b1 = ldB_bf(lsW + (col0 + 16 + lr) * LDK, lane);
    acc[0][0] = wmma_bf16(a0, b0, acc[0][0]);
    acc[0][1] = wmma_bf16(a0, b1, acc[0][1]);
    acc[1][0] = wmma_bf16(a1, b0, acc[1][0]);
    acc[1][1] = wmma_bf16(a1, b1, acc[1][1]);
  }

  const unsigned short* tx0 = ta + (size_t)(rowblk + row0 + lr) * AR;
  const unsigned short* tx1 = ta + (size_t)(rowblk + row0 + 16 + lr) * AR;
  const unsigned short* db0 = dwbt + (size_t)(colblk + col0 + lr) * AR;
  const unsigned short* db1 = dwbt + (size_t)(colblk + col0 + 16 + lr) * AR;
#pragma unroll
  for (int k = 0; k < AR; k += 32) {
    v16bf a0 = ldA(tx0 + k, lane);
    v16bf a1 = ldA(tx1 + k, lane);
    v16bf b0 = ldB_bf(db0 + k, lane);
    v16bf b1 = ldB_bf(db1 + k, lane);
    acc[0][0] = wmma_bf16(a0, b0, acc[0][0]);
    acc[0][1] = wmma_bf16(a0, b1, acc[0][1]);
    acc[1][0] = wmma_bf16(a1, b0, acc[1][0]);
    acc[1][1] = wmma_bf16(a1, b1, acc[1][1]);
  }

  const int rbase = (lane >> 4) << 3;
#pragma unroll
  for (int mt = 0; mt < 2; mt++)
#pragma unroll
    for (int nt = 0; nt < 2; nt++)
#pragma unroll
      for (int v = 0; v < 8; v++) {
        int rr = rowblk + row0 + mt * 16 + rbase + v;
        int cc = colblk + col0 + nt * 16 + lr;
        out[(size_t)rr * HID + cc] = acc[mt][nt][v];
      }
}

// ---------------- launch ----------------

extern "C" void kernel_launch(void* const* d_in, const int* in_sizes, int n_in,
                              void* d_out, int out_size, void* d_ws, size_t ws_size,
                              hipStream_t stream) {
  (void)in_sizes; (void)n_in; (void)out_size; (void)ws_size;

  const float* x       = (const float*)d_in[0];
  const float* gate_w  = (const float*)d_in[1];
  const float* up_w    = (const float*)d_in[2];
  const float* down_w  = (const float*)d_in[3];
  const float* gate_wa = (const float*)d_in[4];
  const float* gate_wb = (const float*)d_in[5];
  const float* up_wa   = (const float*)d_in[6];
  const float* up_wb   = (const float*)d_in[7];
  const float* down_wa = (const float*)d_in[8];
  const float* down_wb = (const float*)d_in[9];
  const int*   seg_ids = (const int*)d_in[10];
  float* out = (float*)d_out;

  // workspace layout (bytes, 256-aligned)
  char* ws = (char*)d_ws;
  const size_t off_xb   = 0;                                   // T*H bf16
  const size_t off_xg   = off_xb   + (size_t)T_TOK * HID * 2;  // T*64 bf16
  const size_t off_xu   = off_xg   + (size_t)T_TOK * AR * 2;
  const size_t off_gwbt = off_xu   + (size_t)T_TOK * AR * 2;   // I*64 bf16
  const size_t off_uwbt = off_gwbt + (size_t)INTER * AR * 2;
  const size_t off_dwbt = off_uwbt + (size_t)INTER * AR * 2;   // H*64 bf16
  const size_t off_t    = off_dwbt + (size_t)HID * AR * 2;     // T*I bf16
  const size_t off_ta   = off_t    + (size_t)T_TOK * INTER * 2;

  unsigned short* xb   = (unsigned short*)(ws + off_xb);
  unsigned short* xg   = (unsigned short*)(ws + off_xg);
  unsigned short* xu   = (unsigned short*)(ws + off_xu);
  unsigned short* gwbt = (unsigned short*)(ws + off_gwbt);
  unsigned short* uwbt = (unsigned short*)(ws + off_uwbt);
  unsigned short* dwbt = (unsigned short*)(ws + off_dwbt);
  unsigned short* tmat = (unsigned short*)(ws + off_t);
  unsigned short* ta   = (unsigned short*)(ws + off_ta);

  cvt_x_kernel<<<(T_TOK * HID) / (256 * 4), 256, 0, stream>>>(x, xb);
  wbt_kernel<<<(INTER * AR) / 256, 256, 0, stream>>>(gate_wb, gwbt, INTER);
  wbt_kernel<<<(INTER * AR) / 256, 256, 0, stream>>>(up_wb, uwbt, INTER);
  wbt_kernel<<<(HID * AR) / 256, 256, 0, stream>>>(down_wb, dwbt, HID);
  lora_proj_gu_kernel<<<T_TOK, 256, 0, stream>>>(x, gate_wa, up_wa, seg_ids, xg, xu);

  gemm_gate_up_kernel<<<dim3(T_TOK / 128, INTER / 64), 256, 0, stream>>>(
      xb, gate_w, up_w, xg, xu, gwbt, uwbt, tmat);

  lora_proj_down_kernel<<<T_TOK, 256, 0, stream>>>(tmat, down_wa, seg_ids, ta);

  gemm_down_kernel<<<dim3(T_TOK / 128, HID / 64), 256, 0, stream>>>(
      tmat, down_w, ta, dwbt, out);
}
